// GNN_ActorCritic_29515015258642
// MI455X (gfx1250) — matrix operand order repaired and checked by reading, not compiled, not run on hardware
//
#include <hip/hip_runtime.h>

// ---------------------------------------------------------------------------
// GNN ActorCritic forward for gfx1250 (MI455X): bf16 WMMA GEMMs with
// async-to-LDS B-panel staging + atomic segment-softmax GAT scatter.
// Wave32 everywhere.
// ---------------------------------------------------------------------------

typedef __attribute__((ext_vector_type(16))) __bf16 v16bf;
typedef __attribute__((ext_vector_type(8)))  __bf16 v8bf;
typedef __attribute__((ext_vector_type(8)))  float  v8f;

union AFrag { v16bf v; v8bf h[2]; };

#define DMODEL 256
#define NGRAPH 8

// ---- helpers ---------------------------------------------------------------

__device__ __forceinline__ unsigned short f2bf(float f) {
  unsigned u = __float_as_uint(f);
  unsigned r = (u + 0x7fffu + ((u >> 16) & 1u)) >> 16;   // round-to-nearest-even
  return (unsigned short)r;
}

// monotonic float <-> uint mapping for atomic max (init value 0u == -inf side)
__device__ __forceinline__ unsigned f2ord(float f) {
  unsigned u = __float_as_uint(f);
  return (u & 0x80000000u) ? ~u : (u | 0x80000000u);
}
__device__ __forceinline__ float ord2f(unsigned s) {
  unsigned u = (s & 0x80000000u) ? (s & 0x7fffffffu) : ~s;
  return __uint_as_float(u);
}

// ---- small utility kernels -------------------------------------------------

__global__ void k_zero_b32(unsigned* __restrict__ p, long n) {
  long t = (long)blockIdx.x * blockDim.x + threadIdx.x;
  if (t < n) p[t] = 0u;
}

__global__ void k_f2bf_arr(const float* __restrict__ s, unsigned short* __restrict__ d, long n) {
  long t = (long)blockIdx.x * blockDim.x + threadIdx.x;
  if (t < n) d[t] = f2bf(s[t]);
}

// Wt[c*K + r] = bf16(W[r*Ncol + c])   (pre-transpose weights for B fragments)
__global__ void k_transpose_bf(const float* __restrict__ W, unsigned short* __restrict__ Wt,
                               int K, int Ncol) {
  int t = blockIdx.x * blockDim.x + threadIdx.x;
  if (t >= K * Ncol) return;
  int r = t / Ncol, c = t - r * Ncol;
  Wt[c * K + r] = f2bf(W[t]);
}

// ---- bf16 WMMA GEMM: C[M,256] = A[M,K](bf16) x Bt[256,K](bf16)^T -----------
// Block = 2 row-tiles x 256 cols (8 waves, 16x64 tile each). The block
// cooperatively stages the 256x32 B panel chunk into LDS with
// GLOBAL_LOAD_ASYNC_TO_LDS_B128 (ASYNCcnt path), then waves feed WMMA B
// fragments from LDS (ds_load_b128: lane reads 32 contiguous bytes).
__global__ void __launch_bounds__(256)
k_gemm_bf16(const unsigned short* __restrict__ A,
            const unsigned short* __restrict__ Bt,
            float* __restrict__ C, int M, int K) {
  __shared__ __align__(64) unsigned short lB[DMODEL * 32];   // 16 KB

  const int tid  = threadIdx.x;
  const int lane = tid & 31;
  const int wave = tid >> 5;
  const int rowTile = blockIdx.x * 2 + (wave >> 2);
  const int colBase = (wave & 3) * 64;
  const bool active = (rowTile * 16 < M);
  const int rowBase = rowTile * 16;
  const int m     = active ? (rowBase + (lane & 15)) : 0;
  const int khalf = lane >> 4;                      // 0|1 (lane half)
  const int nb    = colBase + (lane & 15);

  const unsigned ldsOff = ((unsigned)(unsigned long long)(&lB[0])) + (unsigned)tid * 64u;
  const unsigned short* gB = Bt + (long)tid * K;    // thread t stages column t

  v8f acc0 = {}, acc1 = {}, acc2 = {}, acc3 = {};
  const unsigned short* arow = A + (long)m * K;

  for (int k = 0; k < K; k += 32) {
    // --- cooperative async stage: 64B (32 k-values) per thread -> LDS ------
    {
      unsigned long long ga = (unsigned long long)(gB + k);
      asm volatile(
        "global_load_async_to_lds_b128 %0, %1, off\n\t"
        "global_load_async_to_lds_b128 %0, %1, off offset:16\n\t"
        "global_load_async_to_lds_b128 %0, %1, off offset:32\n\t"
        "global_load_async_to_lds_b128 %0, %1, off offset:48\n\t"
        "s_wait_asynccnt 0"
        :: "v"(ldsOff), "v"(ga) : "memory");
    }
    __syncthreads();

    if (active) {
      AFrag a;                                      // A 16x32, contiguous k order: k = 16*khalf + s
      const unsigned short* ap = arow + k + khalf * 16;
      a.h[0] = *(const v8bf*)(ap);                  // K = k + khalf*16 + [0..7]
      a.h[1] = *(const v8bf*)(ap + 8);              // K = k + khalf*16 + 8 + [0..7]
      if (k + 32 < K) __builtin_prefetch(arow + k + 32, 0, 1);  // global_prefetch_b8

      // B fragments from LDS: col c lives at lB[c*32 .. c*32+31]
      const unsigned short* lb = &lB[khalf * 16];
      v16bf b0 = *(const v16bf*)(lb + (nb +  0) * 32);
      v16bf b1 = *(const v16bf*)(lb + (nb + 16) * 32);
      v16bf b2 = *(const v16bf*)(lb + (nb + 32) * 32);
      v16bf b3 = *(const v16bf*)(lb + (nb + 48) * 32);

      acc0 = __builtin_amdgcn_wmma_f32_16x16x32_bf16(false, a.v, false, b0, (short)0, acc0, false, false);
      acc1 = __builtin_amdgcn_wmma_f32_16x16x32_bf16(false, a.v, false, b1, (short)0, acc1, false, false);
      acc2 = __builtin_amdgcn_wmma_f32_16x16x32_bf16(false, a.v, false, b2, (short)0, acc2, false, false);
      acc3 = __builtin_amdgcn_wmma_f32_16x16x32_bf16(false, a.v, false, b3, (short)0, acc3, false, false);
    }
    __syncthreads();
  }

  if (active) {
    // C/D layout: VGPR i -> row (i + lanehalf*8), col = lane&15
    const int col = colBase + (lane & 15);
#pragma unroll
    for (int i = 0; i < 8; ++i) {
      float* cr = C + (long)(rowBase + i + khalf * 8) * DMODEL + col;
      cr[0] = acc0[i]; cr[16] = acc1[i]; cr[32] = acc2[i]; cr[48] = acc3[i];
    }
  }
}

// ---- GAT attention coefficients: al[n,h] = sum_c h[n,h,c]*a[h,c] -----------
__global__ void k_att_coeff(const float* __restrict__ h,
                            const float* __restrict__ as_, const float* __restrict__ ad_,
                            float* __restrict__ al_s, float* __restrict__ al_d,
                            int N, int H) {
  int wv   = (blockIdx.x * blockDim.x + threadIdx.x) >> 5;   // one wave per node
  int lane = threadIdx.x & 31;
  if (wv >= N) return;
  const float* hp = h + (long)wv * DMODEL + lane * 8;
  float ss = 0.f, sd = 0.f;
#pragma unroll
  for (int i = 0; i < 8; ++i) {
    float v = hp[i];
    ss += v * as_[lane * 8 + i];
    sd += v * ad_[lane * 8 + i];
  }
  int g = 32 / H;                                  // lanes per head (8 or 32)
  for (int off = g >> 1; off; off >>= 1) {
    ss += __shfl_xor(ss, off, 32);
    sd += __shfl_xor(sd, off, 32);
  }
  if ((lane & (g - 1)) == 0) {
    int hd = lane / g;
    al_s[wv * H + hd] = ss;
    al_d[wv * H + hd] = sd;
  }
}

// ---- edge pass 1: segment max of leaky-relu scores -------------------------
__global__ void k_edge_max(const int* __restrict__ ei,
                           const float* __restrict__ al_s, const float* __restrict__ al_d,
                           unsigned* __restrict__ emax, int E, int Etot, int H) {
  int t = blockIdx.x * blockDim.x + threadIdx.x;
  if (t >= Etot * H) return;
  int e = t / H, hd = t - e * H;
  int s = (e < E) ? ei[e] : (e - E);               // self-loops appended
  int d = (e < E) ? ei[E + e] : (e - E);
  float v = al_s[s * H + hd] + al_d[d * H + hd];
  v = (v > 0.f) ? v : 0.2f * v;                    // leaky_relu(0.2)
  atomicMax(emax + d * H + hd, f2ord(v));
}

// ---- edge pass 2: w = exp(e - max), denom = segment sum --------------------
__global__ void k_edge_weight(const int* __restrict__ ei,
                              const float* __restrict__ al_s, const float* __restrict__ al_d,
                              const unsigned* __restrict__ emax,
                              float* __restrict__ wbuf, float* __restrict__ denom,
                              int E, int Etot, int H) {
  int t = blockIdx.x * blockDim.x + threadIdx.x;
  if (t >= Etot * H) return;
  int e = t / H, hd = t - e * H;
  int s = (e < E) ? ei[e] : (e - E);
  int d = (e < E) ? ei[E + e] : (e - E);
  float v = al_s[s * H + hd] + al_d[d * H + hd];
  v = (v > 0.f) ? v : 0.2f * v;
  float w = __expf(v - ord2f(emax[d * H + hd]));
  wbuf[t] = w;
  atomicAdd(denom + d * H + hd, w);
}

// ---- edge pass 3: agg[dst] += alpha * h[src]  (wave per edge) --------------
__global__ void __launch_bounds__(256)
k_edge_scatter(const int* __restrict__ ei, const float* __restrict__ wbuf,
               const float* __restrict__ denom, const float* __restrict__ h,
               float* __restrict__ agg, int E, int Etot, int H) {
  int e    = (blockIdx.x * blockDim.x + threadIdx.x) >> 5;
  int lane = threadIdx.x & 31;
  if (e >= Etot) return;
  int s = (e < E) ? ei[e] : (e - E);
  int d = (e < E) ? ei[E + e] : (e - E);
  int hd = (lane * 8) / (DMODEL / H);              // head of this lane's 8 features
  float alpha = wbuf[e * H + hd] / (denom[d * H + hd] + 1e-16f);
  const float* hp = h + (long)s * DMODEL + lane * 8;
  float* op       = agg + (long)d * DMODEL + lane * 8;
#pragma unroll
  for (int i = 0; i < 8; ++i) atomicAdd(op + i, hp[i] * alpha);
}

// ---- BatchNorm stats (sum / sumsq per feature) -----------------------------
__global__ void k_bn_stats(const float* __restrict__ h, float* __restrict__ stats, int N) {
  int f = threadIdx.x;                             // 256 features
  int chunk = (N + gridDim.x - 1) / gridDim.x;
  int n0 = blockIdx.x * chunk;
  int n1 = n0 + chunk; if (n1 > N) n1 = N;
  float s = 0.f, s2 = 0.f;
  for (int n = n0; n < n1; ++n) {
    float v = h[(long)n * DMODEL + f];
    s += v; s2 += v * v;
  }
  atomicAdd(stats + f, s);
  atomicAdd(stats + DMODEL + f, s2);
}

// ---- BN normalize + ReLU, emit f32 + bf16 ----------------------------------
// (GAT bias cancels exactly through BN: (x+b) - mean(x+b) == x - mean(x))
__global__ void k_bn_relu(const float* __restrict__ agg, const float* __restrict__ stats,
                          const float* __restrict__ gamma, const float* __restrict__ beta,
                          float* __restrict__ outF, unsigned short* __restrict__ outB, int N) {
  long t = (long)blockIdx.x * blockDim.x + threadIdx.x;
  if (t >= (long)N * DMODEL) return;
  int f = (int)(t & (DMODEL - 1));
  float inv = 1.f / (float)N;
  float m = stats[f] * inv;
  float v = stats[DMODEL + f] * inv - m * m;
  float r = (agg[t] - m) * rsqrtf(v + 1e-5f) * gamma[f] + beta[f];
  r = (r > 0.f) ? r : 0.f;
  outF[t] = r;
  outB[t] = f2bf(r);
}

// ---- global mean pool ------------------------------------------------------
__global__ void k_pool(const float* __restrict__ xn, const int* __restrict__ batch,
                       float* __restrict__ sums, float* __restrict__ cnt, int N) {
  long t = (long)blockIdx.x * blockDim.x + threadIdx.x;
  if (t >= (long)N * DMODEL) return;
  int n = (int)(t >> 8), f = (int)(t & (DMODEL - 1));
  int b = batch[n];
  atomicAdd(sums + b * DMODEL + f, xn[t]);
  if (f == 0) atomicAdd(cnt + b, 1.f);
}

// ---- shared = relu(global_feat @ W_sh + b_sh)  (one block) -----------------
__global__ void __launch_bounds__(256)
k_shared(const float* __restrict__ sums, const float* __restrict__ cnt,
         const float* __restrict__ Wsh, const float* __restrict__ bsh,
         float* __restrict__ sh) {
  __shared__ float gf[NGRAPH * DMODEL];
  int j = threadIdx.x;
  for (int b = 0; b < NGRAPH; ++b) {
    float c = cnt[b]; c = (c > 1.f) ? c : 1.f;
    gf[b * DMODEL + j] = sums[b * DMODEL + j] / c;
  }
  __syncthreads();
  for (int b = 0; b < NGRAPH; ++b) {
    float acc = bsh[j];
    for (int k = 0; k < DMODEL; ++k) acc += gf[b * DMODEL + k] * Wsh[k * DMODEL + j];
    sh[b * DMODEL + j] = (acc > 0.f) ? acc : 0.f;
  }
}

// ---- node scores: [x_nodes, shared[batch]] @ W_node + b  (wave per node) ---
__global__ void k_node_scores(const float* __restrict__ xn, const float* __restrict__ sh,
                              const int* __restrict__ batch, const float* __restrict__ Wn,
                              const float* __restrict__ bn, float* __restrict__ out, int N) {
  int n    = (blockIdx.x * blockDim.x + threadIdx.x) >> 5;
  int lane = threadIdx.x & 31;
  if (n >= N) return;
  int b = batch[n];
  float acc = 0.f;
  int base = lane * 16;
#pragma unroll
  for (int i = 0; i < 16; ++i) {
    int k = base + i;
    float v = (k < DMODEL) ? xn[(long)n * DMODEL + k] : sh[b * DMODEL + (k - DMODEL)];
    acc += v * Wn[k];
  }
  for (int off = 16; off; off >>= 1) acc += __shfl_xor(acc, off, 32);
  if (lane == 0) out[n] = acc + bn[0];
}

// ---- type logits + value (tiny, one block) ---------------------------------
__global__ void k_heads(const float* __restrict__ sh, const float* __restrict__ xn,
                        const int* __restrict__ tgt,
                        const float* __restrict__ Wt, const float* __restrict__ bt,
                        const float* __restrict__ Wc, const float* __restrict__ bc,
                        float* __restrict__ out) {
  int t = threadIdx.x;
  if (t < NGRAPH * 4) {
    int b = t >> 2, a = t & 3;
    int tn = tgt[b];
    float acc = bt[a];
    for (int k = 0; k < 2 * DMODEL; ++k) {
      float v = (k < DMODEL) ? sh[b * DMODEL + k] : xn[(long)tn * DMODEL + (k - DMODEL)];
      acc += v * Wt[k * 4 + a];
    }
    out[b * 4 + a] = acc;                                  // type_logits
  } else if (t < NGRAPH * 4 + NGRAPH) {
    int b = t - NGRAPH * 4;
    float acc = bc[0];
    for (int k = 0; k < DMODEL; ++k) acc += sh[b * DMODEL + k] * Wc[k];
    out[NGRAPH * 4 + b] = acc;                             // value
  }
}

// ---------------------------------------------------------------------------

extern "C" void kernel_launch(void* const* d_in, const int* in_sizes, int n_in,
                              void* d_out, int out_size, void* d_ws, size_t ws_size,
                              hipStream_t stream) {
  const float* x      = (const float*)d_in[0];
  const int*   ei     = (const int*)d_in[1];
  const int*   batch  = (const int*)d_in[2];
  const int*   tgt    = (const int*)d_in[3];
  const float* W1     = (const float*)d_in[4];
  const float* a1s    = (const float*)d_in[5];
  const float* a1d    = (const float*)d_in[6];
  const float* W2     = (const float*)d_in[8];
  const float* a2s    = (const float*)d_in[9];
  const float* a2d    = (const float*)d_in[10];
  const float* g1     = (const float*)d_in[12];
  const float* be1    = (const float*)d_in[13];
  const float* g2     = (const float*)d_in[14];
  const float* be2    = (const float*)d_in[15];
  const float* Wsh    = (const float*)d_in[16];
  const float* bsh    = (const float*)d_in[17];
  const float* Wn     = (const float*)d_in[18];
  const float* bnb    = (const float*)d_in[19];
  const float* Wty    = (const float*)d_in[20];
  const float* bty    = (const float*)d_in[21];
  const float* Wc     = (const float*)d_in[22];
  const float* bc     = (const float*)d_in[23];
  float* out = (float*)d_out;

  const int N    = in_sizes[0] / 64;       // 50000
  const int F    = 64;
  const int E    = in_sizes[1] / 2;        // 400000
  const int Etot = E + N;                  // + self-loops

  // workspace carving
  char* ws = (char*)d_ws;
  auto carve = [&](size_t bytes) { char* p = ws; ws += (bytes + 255) & ~(size_t)255; return p; };
  unsigned short* xb   = (unsigned short*)carve((size_t)N * F * 2);
  unsigned short* W1t  = (unsigned short*)carve((size_t)DMODEL * F * 2);
  unsigned short* W2t  = (unsigned short*)carve((size_t)DMODEL * DMODEL * 2);
  float*          h    = (float*)carve((size_t)N * DMODEL * 4);   // GEMM out (both layers)
  unsigned short* hb   = (unsigned short*)carve((size_t)N * DMODEL * 2);
  float*          xn   = (float*)carve((size_t)N * DMODEL * 4);   // BN/ReLU f32 out
  float*          agg  = (float*)carve((size_t)N * DMODEL * 4);
  float*          al_s = (float*)carve((size_t)N * 4 * 4);
  float*          al_d = (float*)carve((size_t)N * 4 * 4);
  unsigned*       emax = (unsigned*)carve((size_t)N * 4 * 4);
  float*          den  = (float*)carve((size_t)N * 4 * 4);
  float*          wbuf = (float*)carve((size_t)Etot * 4 * 4);
  float*          stat = (float*)carve(2 * DMODEL * 4);
  float*          psum = (float*)carve(NGRAPH * DMODEL * 4);
  float*          pcnt = (float*)carve(NGRAPH * 4);
  float*          shb  = (float*)carve(NGRAPH * DMODEL * 4);

  auto cdiv = [](long a, long b) { return (int)((a + b - 1) / b); };

  // ---- weight / input conversion
  k_f2bf_arr<<<cdiv((long)N * F, 256), 256, 0, stream>>>(x, xb, (long)N * F);
  k_transpose_bf<<<cdiv(F * DMODEL, 256), 256, 0, stream>>>(W1, W1t, F, DMODEL);
  k_transpose_bf<<<cdiv(DMODEL * DMODEL, 256), 256, 0, stream>>>(W2, W2t, DMODEL, DMODEL);

  const int gemmBlocks = cdiv(cdiv(N, 16), 2);     // 2 row-tiles per block
  const long ND = (long)N * DMODEL;

  for (int layer = 0; layer < 2; ++layer) {
    const int H = (layer == 0) ? 4 : 1;
    const float* as_ = (layer == 0) ? a1s : a2s;
    const float* ad_ = (layer == 0) ? a1d : a2d;
    const float* gm  = (layer == 0) ? g1 : g2;
    const float* bt  = (layer == 0) ? be1 : be2;

    // linear: h = A @ W  (bf16 WMMA, f32 accumulate, async-LDS B staging)
    if (layer == 0)
      k_gemm_bf16<<<gemmBlocks, 256, 0, stream>>>(xb, W1t, h, N, F);
    else
      k_gemm_bf16<<<gemmBlocks, 256, 0, stream>>>(hb, W2t, h, N, DMODEL);

    // zero accumulators
    k_zero_b32<<<cdiv((long)N * H, 256), 256, 0, stream>>>(emax, (long)N * H);
    k_zero_b32<<<cdiv((long)N * H, 256), 256, 0, stream>>>((unsigned*)den, (long)N * H);
    k_zero_b32<<<cdiv(ND, 256), 256, 0, stream>>>((unsigned*)agg, ND);
    k_zero_b32<<<1, 256, 0, stream>>>((unsigned*)stat, 2 * DMODEL);

    // attention
    k_att_coeff<<<cdiv((long)N * 32, 256), 256, 0, stream>>>(h, as_, ad_, al_s, al_d, N, H);
    k_edge_max<<<cdiv((long)Etot * H, 256), 256, 0, stream>>>(ei, al_s, al_d, emax, E, Etot, H);
    k_edge_weight<<<cdiv((long)Etot * H, 256), 256, 0, stream>>>(ei, al_s, al_d, emax, wbuf, den, E, Etot, H);
    k_edge_scatter<<<cdiv((long)Etot * 32, 256), 256, 0, stream>>>(ei, wbuf, den, h, agg, E, Etot, H);

    // BN + ReLU (+ bf16 copy for next GEMM)
    k_bn_stats<<<256, 256, 0, stream>>>(agg, stat, N);
    k_bn_relu<<<cdiv(ND, 256), 256, 0, stream>>>(agg, stat, gm, bt, xn, hb, N);
  }

  // ---- pooling + heads
  k_zero_b32<<<cdiv((long)NGRAPH * DMODEL + NGRAPH, 256), 256, 0, stream>>>((unsigned*)psum, NGRAPH * DMODEL);
  k_zero_b32<<<1, 32, 0, stream>>>((unsigned*)pcnt, NGRAPH);
  k_pool<<<cdiv(ND, 256), 256, 0, stream>>>(xn, batch, psum, pcnt, N);
  k_shared<<<1, 256, 0, stream>>>(psum, pcnt, Wsh, bsh, shb);
  k_node_scores<<<cdiv((long)N * 32, 256), 256, 0, stream>>>(xn, shb, batch, Wn, bnb, out, N);
  k_heads<<<1, 64, 0, stream>>>(shb, xn, tgt, Wty, bty, Wc, bc, out + N);
}
